// MSAPairUpdateBlock_7164005450353
// MI455X (gfx1250) — matrix operand order repaired and checked by reading, not compiled, not run on hardware
//
#include <hip/hip_runtime.h>
#include <hip/hip_bf16.h>
#include <math.h>

// MSA pair-update block for gfx1250 (MI455X), wave32 + WMMA bf16 + TDM.
// f32 residual stream in HBM; bf16 intermediates (channel-major where the
// consumer needs contiguous-K); all tiles staged with vector loads/stores or
// the Tensor Data Mover; all fragments are 16B/32B vector reads.

typedef __attribute__((ext_vector_type(2)))  __bf16 bf16x2;
typedef __attribute__((ext_vector_type(4)))  __bf16 bf16x4;
typedef __attribute__((ext_vector_type(8)))  __bf16 bf16x8;
typedef __attribute__((ext_vector_type(16))) __bf16 bf16x16;
typedef __attribute__((ext_vector_type(2)))  float  f32x2;
typedef __attribute__((ext_vector_type(4)))  float  f32x4;
typedef __attribute__((ext_vector_type(8)))  float  f32x8;
typedef __attribute__((ext_vector_type(4)))  unsigned int u32x4;
typedef __attribute__((ext_vector_type(4)))  int    i32x4;
typedef __attribute__((ext_vector_type(8)))  int    i32x8;

#define NRES   256
#define CDIM   128
#define TOKENS (NRES * NRES)   // 65536

#if __has_builtin(__builtin_amdgcn_tensor_load_to_lds)
#define HAVE_TDM 1
#else
#define HAVE_TDM 0
#endif

// ---- WMMA fragment maps (wave32, 16x16x32 bf16) ----------------------------
// A (16x32): lane l: m=l&15, half=l>>4; elems are K runs [8h..8h+7],[16+8h..+7]
// B (32x16): lane l: n=l&15, half=l>>4; elems are K runs [8h..8h+7],[16+8h..+7]
// C/D:       lane l: n=l&15, half=l>>4; reg r -> m = r + 8*half
__device__ __forceinline__ bf16x16 join8(bf16x8 lo, bf16x8 hi) {
  union { bf16x16 v; bf16x8 h[2]; } u;
  u.h[0] = lo; u.h[1] = hi;
  return u.v;
}
// row = pointer to a 32-element K-run for this lane's row
__device__ __forceinline__ bf16x16 fragA(const __bf16* row, int half) {
  return join8(*(const bf16x8*)(row + 8 * half),
               *(const bf16x8*)(row + 16 + 8 * half));
}
__device__ __forceinline__ bf16x16 fragB(const __bf16* row, int half) {
  return join8(*(const bf16x8*)(row + 8 * half),
               *(const bf16x8*)(row + 16 + 8 * half));
}

__device__ __forceinline__ f32x8 wmma_bf16(bf16x16 a, bf16x16 b, f32x8 c) {
  return __builtin_amdgcn_wmma_f32_16x16x32_bf16(false, a, false, b,
                                                 (short)0, c, false, false);
}

__device__ __forceinline__ float sigm(float x) {
  return 1.0f / (1.0f + __expf(-x));
}

#define ZERO8 {0.f,0.f,0.f,0.f,0.f,0.f,0.f,0.f}

enum { OUT_F32 = 0, OUT_BF16_TM = 1, OUT_BF16_CM = 2 };

// ---- Shared staging helpers ------------------------------------------------
// 128x32 f32 tile -> bf16 LDS [row][32]; float4 global loads, b64 LDS stores.
__device__ __forceinline__ void stage_a_f32(const float* __restrict__ A, size_t row0,
                                            int K, int k0, __bf16* lds, int tid) {
  f32x4 t[4];
#pragma unroll
  for (int e = 0; e < 4; ++e) {
    int idx4 = e * 256 + tid;                  // 1024 float4
    int r = idx4 >> 3, kb = (idx4 & 7) * 4;
    t[e] = *(const f32x4*)&A[(row0 + r) * (size_t)K + k0 + kb];
  }
#pragma unroll
  for (int e = 0; e < 4; ++e) {
    int idx4 = e * 256 + tid;
    int r = idx4 >> 3, kb = (idx4 & 7) * 4;
    bf16x4 pk;
#pragma unroll
    for (int j = 0; j < 4; ++j) pk[j] = (__bf16)t[e][j];
    *(bf16x4*)&lds[r * 32 + kb] = pk;
  }
}

// 32x16 W tile -> bf16 LDS [n][32] (transposed for B-fragments).
__device__ __forceinline__ void stage_w_f32(const float* __restrict__ W, int N,
                                            int k0, int n0, __bf16* lds, int tid) {
  int r = tid >> 3, c2 = (tid & 7) * 2, nc = n0 + c2;
  float x = 0.f, y = 0.f;
  if (nc + 1 < N) {
    f32x2 w2 = *(const f32x2*)&W[(size_t)(k0 + r) * N + nc];
    x = w2[0]; y = w2[1];
  } else if (nc < N) {
    x = W[(size_t)(k0 + r) * N + nc];
  }
  lds[c2 * 32 + r]       = (__bf16)x;
  lds[(c2 + 1) * 32 + r] = (__bf16)y;
}

#if HAVE_TDM
// ---- Tensor Data Mover: 2D tile (bf16) global -> LDS -----------------------
// D# per CDNA5 ISA. lds_off is a byte offset into the
// workgroup LDS allocation; tiles are tile1 rows x tile0 contiguous elements.
__device__ __forceinline__ void tdm_load_2d_bf16(unsigned lds_off, const void* gaddr,
                                                 unsigned tile0, unsigned tile1,
                                                 unsigned stride0,
                                                 unsigned tdim0, unsigned tdim1) {
  unsigned long long ga = (unsigned long long)(uintptr_t)gaddr;
  u32x4 g0;
  g0[0] = 1u;                                           // count=1 (user D#)
  g0[1] = lds_off;                                      // lds_addr (bytes)
  g0[2] = (unsigned)ga;                                 // global_addr[31:0]
  g0[3] = (unsigned)((ga >> 32) & 0x01FFFFFFu) | 0x80000000u; // [56:32]|type=2
  i32x8 g1;
  g1[0] = (int)(1u << 16);                              // data_size=1 (2 bytes)
  g1[1] = (int)((tdim0 & 0xFFFFu) << 16);               // tensor_dim0 lo
  g1[2] = (int)((tdim0 >> 16) | ((tdim1 & 0xFFFFu) << 16));
  g1[3] = (int)((tdim1 >> 16) | (tile0 << 16));         // tile_dim0
  g1[4] = (int)(tile1 & 0xFFFFu);                       // tile_dim1, tile_dim2=0
  g1[5] = (int)stride0;                                 // tensor_dim0_stride lo
  g1[6] = 0;
  g1[7] = 0;
  i32x4 z4 = {0, 0, 0, 0};
  i32x8 z8 = {0, 0, 0, 0, 0, 0, 0, 0};
  // 6-arg form on this toolchain: (g0, g1, g2, g3, g4, cpol)
  __builtin_amdgcn_tensor_load_to_lds(g0, g1, z4, z4, z8, 0);
}
#endif

// ---- LayerNorm over last dim (128), one wave per token ---------------------
__global__ void ln_kernel(const float* __restrict__ x, const float* __restrict__ g,
                          const float* __restrict__ b, float* __restrict__ out) {
  const int wv = threadIdx.x >> 5, lane = threadIdx.x & 31;
  const size_t t = (size_t)blockIdx.x * 8 + wv;
  const int c4 = lane * 4;
  f32x4 v = *(const f32x4*)&x[t * CDIM + c4];
  float s = v[0] + v[1] + v[2] + v[3];
  float s2 = v[0]*v[0] + v[1]*v[1] + v[2]*v[2] + v[3]*v[3];
#pragma unroll
  for (int off = 16; off >= 1; off >>= 1) {
    s  += __shfl_xor(s,  off);
    s2 += __shfl_xor(s2, off);
  }
  const float m   = s * (1.0f / CDIM);
  const float var = s2 * (1.0f / CDIM) - m * m;
  const float inv = rsqrtf(var + 1e-5f);
  f32x4 gg = *(const f32x4*)&g[c4];
  f32x4 bb = *(const f32x4*)&b[c4];
  f32x4 o;
#pragma unroll
  for (int e = 0; e < 4; ++e) o[e] = (v[e] - m) * inv * gg[e] + bb[e];
  *(f32x4*)&out[t * CDIM + c4] = o;
}

// ---- Generic token GEMM: A[T,K](f32) @ W[K,N](f32) -------------------------
// grid (T/128, ceil(N/16)), block 256 (8 waves).
__global__ void gemm_kernel(const float* __restrict__ A, const float* __restrict__ W,
                            const float* __restrict__ bias,
                            const float* __restrict__ residual,
                            void* __restrict__ outp,
                            int K, int N, int act, int trans_store, int mode) {
  __shared__ __bf16 lds_a[128 * 32];
  __shared__ __bf16 lds_w[16 * 32];
  __shared__ float  lds_out[128 * 16];
  const int tid = threadIdx.x;
  const int wv = tid >> 5, lane = tid & 31;
  const int half = lane >> 4, mn = lane & 15;
  const int m_blk = blockIdx.x * 128;
  const int n0 = blockIdx.y * 16;

  f32x8 acc = ZERO8;
  for (int k0 = 0; k0 < K; k0 += 32) {
    stage_a_f32(A, (size_t)m_blk, K, k0, lds_a, tid);
    stage_w_f32(W, N, k0, n0, lds_w, tid);
    __syncthreads();
    bf16x16 af  = fragA(&lds_a[(wv * 16 + mn) * 32], half);
    bf16x16 bfr = fragB(&lds_w[mn * 32], half);
    acc = wmma_bf16(af, bfr, acc);
    __syncthreads();
  }

  const int n = n0 + mn;
  const float bv = (bias && n < N) ? bias[n] : 0.0f;
  float vals[8];
#pragma unroll
  for (int r = 0; r < 8; ++r) {
    float v = acc[r] + bv;
    if (act == 1) v = fmaxf(v, 0.0f);
    else if (act == 2) v = sigm(v);
    vals[r] = v;
  }

  if (mode == OUT_BF16_CM) {                 // per-lane 16B vector store
    if (n < N) {
      bf16x8 pk;
#pragma unroll
      for (int r = 0; r < 8; ++r) pk[r] = (__bf16)vals[r];
      __bf16* cm = (__bf16*)outp;
      *(bf16x8*)&cm[(size_t)n * TOKENS + m_blk + wv * 16 + 8 * half] = pk;
    }
    return;
  }

  // stage tile through LDS; coalesced float4 row stores
#pragma unroll
  for (int r = 0; r < 8; ++r)
    lds_out[(wv * 16 + r + 8 * half) * 16 + mn] = vals[r];
  __syncthreads();
#pragma unroll
  for (int e = 0; e < 2; ++e) {
    int idx4 = e * 256 + tid;                // 512 float4 = 128x16
    int m = idx4 >> 2, c4 = (idx4 & 3) * 4;
    int nn = n0 + c4;
    if (nn < N) {
      int t = m_blk + m;
      f32x4 vv = *(const f32x4*)&lds_out[m * 16 + c4];
      if (mode == OUT_BF16_TM) {
        bf16x4 pk;
#pragma unroll
        for (int j = 0; j < 4; ++j) pk[j] = (__bf16)vv[j];
        *(bf16x4*)&((__bf16*)outp)[(size_t)t * N + nn] = pk;
      } else {
        int tt = trans_store ? ((t & 255) * 256 + (t >> 8)) : t;
        if (residual) {
          f32x4 rr = *(const f32x4*)&residual[(size_t)tt * N + nn];
#pragma unroll
          for (int j = 0; j < 4; ++j) vv[j] += rr[j];
        }
        *(f32x4*)&((float*)outp)[(size_t)tt * N + nn] = vv;
      }
    }
  }
}

// ---- Gated projection: out_cm[n][t] = bf16(sigmoid(A@Wg+bg) * (A@Wp+bp)) ---
__global__ void dual_gemm_sigmul_kernel(const float* __restrict__ A,
                                        const float* __restrict__ Wg, const float* __restrict__ bg,
                                        const float* __restrict__ Wp, const float* __restrict__ bp,
                                        __bf16* __restrict__ out_cm) {
  __shared__ __bf16 lds_a[128 * 32];
  __shared__ __bf16 lds_wg[16 * 32];
  __shared__ __bf16 lds_wp[16 * 32];
  const int tid = threadIdx.x;
  const int wv = tid >> 5, lane = tid & 31;
  const int half = lane >> 4, mn = lane & 15;
  const int m_blk = blockIdx.x * 128;
  const int n0 = blockIdx.y * 16;

  f32x8 accg = ZERO8, accp = ZERO8;
  for (int k0 = 0; k0 < CDIM; k0 += 32) {
    stage_a_f32(A, (size_t)m_blk, CDIM, k0, lds_a, tid);
    stage_w_f32(Wg, CDIM, k0, n0, lds_wg, tid);
    stage_w_f32(Wp, CDIM, k0, n0, lds_wp, tid);
    __syncthreads();
    bf16x16 af  = fragA(&lds_a[(wv * 16 + mn) * 32], half);
    bf16x16 bg_ = fragB(&lds_wg[mn * 32], half);
    bf16x16 bp_ = fragB(&lds_wp[mn * 32], half);
    accg = wmma_bf16(af, bg_, accg);         // shared A fragment, two B's
    accp = wmma_bf16(af, bp_, accp);
    __syncthreads();
  }
  const int n = n0 + mn;
  bf16x8 pk;
#pragma unroll
  for (int r = 0; r < 8; ++r)
    pk[r] = (__bf16)(sigm(accg[r] + bg[n]) * (accp[r] + bp[n]));
  *(bf16x8*)&out_cm[(size_t)n * TOKENS + m_blk + wv * 16 + 8 * half] = pk;
}

// ---- Fused tri-mul output: z = z + sigmoid(zn@Wg+bg) * (xn@Wo+bo) ----------
__global__ void fused_gate_out_kernel(const float* __restrict__ ZN,
                                      const float* __restrict__ Wg, const float* __restrict__ bg,
                                      const float* __restrict__ XN,
                                      const float* __restrict__ Wo, const float* __restrict__ bo,
                                      const float* __restrict__ residual,
                                      float* __restrict__ out) {
  __shared__ __bf16 lds_a1[128 * 32];
  __shared__ __bf16 lds_a2[128 * 32];
  __shared__ __bf16 lds_w1[16 * 32];
  __shared__ __bf16 lds_w2[16 * 32];
  __shared__ float  lds_out[128 * 16];
  const int tid = threadIdx.x;
  const int wv = tid >> 5, lane = tid & 31;
  const int half = lane >> 4, mn = lane & 15;
  const int m_blk = blockIdx.x * 128;
  const int n0 = blockIdx.y * 16;

  f32x8 acc1 = ZERO8, acc2 = ZERO8;
  for (int k0 = 0; k0 < CDIM; k0 += 32) {
    stage_a_f32(ZN, (size_t)m_blk, CDIM, k0, lds_a1, tid);
    stage_a_f32(XN, (size_t)m_blk, CDIM, k0, lds_a2, tid);
    stage_w_f32(Wg, CDIM, k0, n0, lds_w1, tid);
    stage_w_f32(Wo, CDIM, k0, n0, lds_w2, tid);
    __syncthreads();
    bf16x16 a1 = fragA(&lds_a1[(wv * 16 + mn) * 32], half);
    bf16x16 a2 = fragA(&lds_a2[(wv * 16 + mn) * 32], half);
    bf16x16 b1 = fragB(&lds_w1[mn * 32], half);
    bf16x16 b2 = fragB(&lds_w2[mn * 32], half);
    acc1 = wmma_bf16(a1, b1, acc1);
    acc2 = wmma_bf16(a2, b2, acc2);
    __syncthreads();
  }
  const int n = n0 + mn;
#pragma unroll
  for (int r = 0; r < 8; ++r)
    lds_out[(wv * 16 + r + 8 * half) * 16 + mn] =
        sigm(acc1[r] + bg[n]) * (acc2[r] + bo[n]);
  __syncthreads();
#pragma unroll
  for (int e = 0; e < 2; ++e) {
    int idx4 = e * 256 + tid;
    int m = idx4 >> 2, c4 = (idx4 & 3) * 4;
    size_t t = (size_t)(m_blk + m);
    f32x4 vv = *(const f32x4*)&lds_out[m * 16 + c4];
    f32x4 rr = *(const f32x4*)&residual[t * CDIM + n0 + c4];
#pragma unroll
    for (int j = 0; j < 4; ++j) vv[j] += rr[j];
    *(f32x4*)&out[t * CDIM + n0 + c4] = vv;
  }
}

// ---- Triangle einsum: per-channel 256x256x256 GEMM on channel-major bf16 ---
// outgoing: X[i,j] = sum_k a[i,k]*b[j,k];  incoming: X[i,j] = sum_k a[k,i]*b[k,j]
// grid (32, 128). Outgoing tiles staged by the Tensor Data Mover.
__global__ void tri_einsum_kernel(const __bf16* __restrict__ Acm,
                                  const __bf16* __restrict__ Bcm,
                                  float* __restrict__ Xcm, int outgoing) {
  __shared__ __align__(16) unsigned char smem[128*32*2 + 16*32*2 + 128*16*4];
  __bf16* lds_a  = (__bf16*)smem;                    // LDS byte offset 0
  __bf16* lds_w  = (__bf16*)(smem + 8192);           // LDS byte offset 8192
  float*  lds_out = (float*)(smem + 9216);
  const int tid = threadIdx.x;
  const int wv = tid >> 5, lane = tid & 31;
  const int half = lane >> 4, mn = lane & 15;
  const int c = blockIdx.y;
  const int i0 = (blockIdx.x >> 4) * 128;
  const int j0 = (blockIdx.x & 15) * 16;
  const __bf16* pa = Acm + (size_t)c * TOKENS;
  const __bf16* pb = Bcm + (size_t)c * TOKENS;

  f32x8 acc = ZERO8;
  for (int k0 = 0; k0 < NRES; k0 += 32) {
    if (outgoing) {
#if HAVE_TDM
      if (wv == 0) {                         // one wave drives the DMA engine
        tdm_load_2d_bf16(0u,    pa + (size_t)i0 * NRES + k0, 32u, 128u,
                         (unsigned)NRES, (unsigned)NRES, (unsigned)TOKENS);
        tdm_load_2d_bf16(8192u, pb + (size_t)j0 * NRES + k0, 32u, 16u,
                         (unsigned)NRES, (unsigned)NRES, (unsigned)TOKENS);
        __builtin_amdgcn_s_wait_tensorcnt(0);
      }
#else
#pragma unroll
      for (int e = 0; e < 2; ++e) {          // A[m][k]=a[i0+m,k0+k], vec both sides
        int idx8 = e * 256 + tid;
        int m = idx8 >> 2, kb = (idx8 & 3) * 8;
        *(bf16x8*)&lds_a[m * 32 + kb] =
            *(const bf16x8*)&pa[(size_t)(i0 + m) * NRES + k0 + kb];
      }
      if (tid < 64) {                        // B[k][n]=b[j0+n,k0+k]
        int nn = tid >> 2, kb = (tid & 3) * 8;
        *(bf16x8*)&lds_w[nn * 32 + kb] =
            *(const bf16x8*)&pb[(size_t)(j0 + nn) * NRES + k0 + kb];
      }
#endif
    } else {
#pragma unroll
      for (int e = 0; e < 2; ++e) {          // A[m][k]=a[k0+k,i0+m], vec along m
        int idx8 = e * 256 + tid;
        int k = idx8 >> 4, mb = (idx8 & 15) * 8;
        bf16x8 v8 = *(const bf16x8*)&pa[(size_t)(k0 + k) * NRES + i0 + mb];
#pragma unroll
        for (int j = 0; j < 8; ++j) lds_a[(mb + j) * 32 + k] = v8[j];
      }
      if (tid < 64) {                        // B[k][n]=b[k0+k,j0+n]
        int k = tid >> 1, nb = (tid & 1) * 8;
        bf16x8 v8 = *(const bf16x8*)&pb[(size_t)(k0 + k) * NRES + j0 + nb];
#pragma unroll
        for (int j = 0; j < 8; ++j) lds_w[(nb + j) * 32 + k] = v8[j];
      }
    }
    __syncthreads();
    bf16x16 af  = fragA(&lds_a[(wv * 16 + mn) * 32], half);
    bf16x16 bfr = fragB(&lds_w[mn * 32], half);
    acc = wmma_bf16(af, bfr, acc);
    __syncthreads();
  }
#pragma unroll
  for (int r = 0; r < 8; ++r)
    lds_out[(wv * 16 + r + 8 * half) * 16 + mn] = acc[r];
  __syncthreads();
#pragma unroll
  for (int e = 0; e < 2; ++e) {
    int idx4 = e * 256 + tid;
    int m = idx4 >> 2, j4 = (idx4 & 3) * 4;
    *(f32x4*)&Xcm[(size_t)c * TOKENS + (size_t)(i0 + m) * NRES + j0 + j4] =
        *(const f32x4*)&lds_out[m * 16 + j4];
  }
}

// ---- Tiled transpose: channel-major [128][65536] f32 -> token-major --------
__global__ void transpose_cm_tm_kernel(const float* __restrict__ in,
                                       float* __restrict__ out) {
  __shared__ float tile[32][36];
  const int tid = threadIdx.x;
  const int tb = blockIdx.x;   // token tile (2048)
  const int cb = blockIdx.y;   // channel tile (4)
  {
    int r = tid >> 3, col4 = (tid & 7) * 4;
    f32x4 v = *(const f32x4*)&in[(size_t)(cb * 32 + r) * TOKENS + tb * 32 + col4];
    *(f32x4*)&tile[r][col4] = v;
  }
  __syncthreads();
  {
    int tr = tid >> 3, cc4 = (tid & 7) * 4;
    f32x4 v;
#pragma unroll
    for (int j = 0; j < 4; ++j) v[j] = tile[cc4 + j][tr];
    *(f32x4*)&out[(size_t)(tb * 32 + tr) * CDIM + cb * 32 + cc4] = v;
  }
}

// ---- Triangle attention: softmax(QK^T*s + bias) @ V ------------------------
// q,k: bf16 token-major [T][128]; v: bf16 channel-major [128][T]; tb: f32.
__global__ void tri_attn_kernel(const __bf16* __restrict__ q, const __bf16* __restrict__ kb,
                                const __bf16* __restrict__ vcm, const float* __restrict__ tb,
                                float* __restrict__ o) {
  __shared__ float  sS[32 * 256];
  __shared__ __bf16 pP[32 * 256];
  const int tid = threadIdx.x;
  const int wv = tid >> 5, lane = tid & 31;
  const int half = lane >> 4, mn = lane & 15;
  const int gx = blockIdx.x;
  const int jb = gx & 7;
  const int h  = (gx >> 3) & 3;
  const int iq = gx >> 5;
  const float scale = 0.17677669529663687f;  // 32^-0.5
  const size_t rowbase = (size_t)iq * NRES;

  // Phase 1: S = QK^T * scale + bias (32 tiles of 16x16; d=32 -> 1 wmma each)
  for (int tIdx = wv; tIdx < 32; tIdx += 8) {
    int tj = tIdx >> 4, tk = tIdx & 15;
    bf16x16 af  = fragA(&q[(rowbase + jb * 32 + tj * 16 + mn) * CDIM + h * 32], half);
    bf16x16 bfr = fragB(&kb[(rowbase + tk * 16 + mn) * CDIM + h * 32], half);
    f32x8 acc = ZERO8;
    acc = wmma_bf16(af, bfr, acc);
#pragma unroll
    for (int r = 0; r < 8; ++r) {
      int sr = tj * 16 + r + 8 * half;
      int kc = tk * 16 + mn;
      int jg = jb * 32 + sr;
      sS[sr * 256 + kc] = acc[r] * scale + tb[((size_t)jg * NRES + kc) * 4 + h];
    }
  }
  __syncthreads();

  // Phase 2: row softmax, 8 lanes per row, shfl_xor reductions; probs -> bf16
  {
    int row = tid >> 3, sub = tid & 7;
    float mx = -3.4e38f;
    for (int cc = sub; cc < 256; cc += 8) mx = fmaxf(mx, sS[row * 256 + cc]);
#pragma unroll
    for (int msk = 4; msk >= 1; msk >>= 1) mx = fmaxf(mx, __shfl_xor(mx, msk));
    float sum = 0.f;
    for (int cc = sub; cc < 256; cc += 8) {
      float ev = __expf(sS[row * 256 + cc] - mx);
      sS[row * 256 + cc] = ev; sum += ev;
    }
#pragma unroll
    for (int msk = 4; msk >= 1; msk >>= 1) sum += __shfl_xor(sum, msk);
    float inv = 1.0f / sum;
    for (int cc = sub; cc < 256; cc += 8)
      pP[row * 256 + cc] = (__bf16)(sS[row * 256 + cc] * inv);
  }
  __syncthreads();

  // Phase 3: O = P @ V (2 j-tiles x 2 d-tiles; K loop 256/32)
  if (wv < 4) {
    int tj = wv >> 1, td = wv & 1;
    const __bf16* vcol = &vcm[(size_t)(h * 32 + td * 16 + mn) * TOKENS + rowbase];
    f32x8 acc = ZERO8;
    for (int k0 = 0; k0 < 256; k0 += 32) {
      bf16x16 af  = fragA(&pP[(tj * 16 + mn) * 256 + k0], half);
      bf16x16 bfr = fragB(&vcol[k0], half);
      acc = wmma_bf16(af, bfr, acc);
    }
#pragma unroll
    for (int r = 0; r < 8; ++r) {
      o[(rowbase + jb * 32 + tj * 16 + r + 8 * half) * CDIM + h * 32 + td * 16 + mn] = acc[r];
    }
  }
}

// ---- Elementwise helpers (float4) ------------------------------------------
__global__ void mul_kernel(const float* __restrict__ a, const float* __restrict__ b,
                           float* __restrict__ out) {
  size_t i4 = ((size_t)blockIdx.x * 256 + threadIdx.x) * 4;
  f32x4 x = *(const f32x4*)&a[i4];
  f32x4 y = *(const f32x4*)&b[i4];
#pragma unroll
  for (int j = 0; j < 4; ++j) x[j] *= y[j];
  *(f32x4*)&out[i4] = x;
}

__global__ void transpose_tok_kernel(const float* __restrict__ in, float* __restrict__ out) {
  size_t idx4 = (size_t)blockIdx.x * 256 + threadIdx.x;
  int c4 = (int)(idx4 & 31) * 4;
  size_t t = idx4 >> 5;
  int i = (int)(t >> 8), j = (int)(t & 255);
  *(f32x4*)&out[((size_t)(j * NRES + i)) * CDIM + c4] =
      *(const f32x4*)&in[t * CDIM + c4];
}

// ---- Input flat indices (JAX pytree = sorted dict keys) --------------------
enum {
  IN_Z = 0,
  PT_L1B, PT_L1W, PT_L2B, PT_L2W, PT_LNB, PT_LNG,
  TAE_BIASW, TAE_GATEB, TAE_GATEW, TAE_KB, TAE_KW, TAE_LNB, TAE_LNG,
  TAE_OUTB, TAE_OUTW, TAE_QB, TAE_QW, TAE_VB, TAE_VW,
  TAS_BIASW, TAS_GATEB, TAS_GATEW, TAS_KB, TAS_KW, TAS_LNB, TAS_LNG,
  TAS_OUTB, TAS_OUTW, TAS_QB, TAS_QW, TAS_VB, TAS_VW,
  TMI_AGB, TMI_AGW, TMI_APB, TMI_APW, TMI_BGB, TMI_BGW, TMI_BPB, TMI_BPW,
  TMI_GATEB, TMI_GATEW, TMI_LNINB, TMI_LNING, TMI_LNOUTB, TMI_LNOUTG,
  TMI_OUTB, TMI_OUTW,
  TMO_AGB, TMO_AGW, TMO_APB, TMO_APW, TMO_BGB, TMO_BGW, TMO_BPB, TMO_BPW,
  TMO_GATEB, TMO_GATEW, TMO_LNINB, TMO_LNING, TMO_LNOUTB, TMO_LNOUTG,
  TMO_OUTB, TMO_OUTW,
};

extern "C" void kernel_launch(void* const* d_in, const int* in_sizes, int n_in,
                              void* d_out, int out_size, void* d_ws, size_t ws_size,
                              hipStream_t stream) {
  (void)in_sizes; (void)n_in; (void)out_size; (void)ws_size;
  auto P = [&](int i) { return (const float*)d_in[i]; };
  float* z = (float*)d_out;              // residual stream, updated in place

  const size_t T = TOKENS;
  float* ws    = (float*)d_ws;
  float* s_zn  = ws;                 // T*128 f32   LN(z)
  float* s_a   = s_zn + T * 128;     // a(cm bf16) / q(tm bf16)
  float* s_b   = s_a  + T * 128;     // b(cm bf16) / k(tm bf16)
  float* s_x   = s_b  + T * 128;     // X(cm f32) / attn O(tm f32)
  float* s_xt  = s_x  + T * 128;     // X(tm f32) / g*o(tm f32)
  float* s_xn  = s_xt + T * 128;     // LN(x) / sigmoid gate
  float* s_g   = s_xn + T * 128;     // v (cm bf16)
  float* s_tb  = s_g  + T * 128;     // T*4 pair bias
  float* s_big = s_tb + T * 4;       // T*512 transition hidden

  const dim3 blk(256);
  const dim3 gG(512, 8);             // token GEMM, N=128
  const int  gLN = TOKENS / 8;
  const int  gEW4 = (TOKENS * 128) / (256 * 4);

  (void)hipMemcpyAsync(z, d_in[IN_Z], (size_t)T * 128 * sizeof(float),
                       hipMemcpyDeviceToDevice, stream);

  // ---- triangle multiplication (outgoing, then incoming) ----
  auto tri_mul_stage = [&](int ag_b, int ag_w, int ap_b, int ap_w,
                           int bg_b, int bg_w, int bp_b, int bp_w,
                           int gate_b, int gate_w, int lnin_b, int lnin_g,
                           int lnout_b, int lnout_g, int out_b, int out_w,
                           int outgoing) {
    ln_kernel<<<gLN, blk, 0, stream>>>(z, P(lnin_g), P(lnin_b), s_zn);
    dual_gemm_sigmul_kernel<<<gG, blk, 0, stream>>>(s_zn, P(ag_w), P(ag_b),
                                                    P(ap_w), P(ap_b), (__bf16*)s_a);
    dual_gemm_sigmul_kernel<<<gG, blk, 0, stream>>>(s_zn, P(bg_w), P(bg_b),
                                                    P(bp_w), P(bp_b), (__bf16*)s_b);
    tri_einsum_kernel<<<dim3(32, 128), blk, 0, stream>>>((const __bf16*)s_a,
                                                         (const __bf16*)s_b, s_x, outgoing);
    transpose_cm_tm_kernel<<<dim3(2048, 4), blk, 0, stream>>>(s_x, s_xt);
    ln_kernel<<<gLN, blk, 0, stream>>>(s_xt, P(lnout_g), P(lnout_b), s_xn);
    fused_gate_out_kernel<<<gG, blk, 0, stream>>>(s_zn, P(gate_w), P(gate_b),
                                                  s_xn, P(out_w), P(out_b), z, z);
  };
  tri_mul_stage(TMO_AGB, TMO_AGW, TMO_APB, TMO_APW, TMO_BGB, TMO_BGW, TMO_BPB, TMO_BPW,
                TMO_GATEB, TMO_GATEW, TMO_LNINB, TMO_LNING, TMO_LNOUTB, TMO_LNOUTG,
                TMO_OUTB, TMO_OUTW, 1);
  tri_mul_stage(TMI_AGB, TMI_AGW, TMI_APB, TMI_APW, TMI_BGB, TMI_BGW, TMI_BPB, TMI_BPW,
                TMI_GATEB, TMI_GATEW, TMI_LNINB, TMI_LNING, TMI_LNOUTB, TMI_LNOUTG,
                TMI_OUTB, TMI_OUTW, 0);

  // ---- triangle attention (starting; ending in transposed space) ----
  auto attn_stage = [&](int ln_b, int ln_g, int q_b, int q_w, int k_b, int k_w,
                        int v_b, int v_w, int bias_w, int gate_b, int gate_w,
                        int out_b, int out_w, int trans) {
    const float* zin = z;
    if (trans) {
      transpose_tok_kernel<<<gEW4, blk, 0, stream>>>(z, s_xt);
      zin = s_xt;
    }
    ln_kernel<<<gLN, blk, 0, stream>>>(zin, P(ln_g), P(ln_b), s_zn);
    gemm_kernel<<<gG, blk, 0, stream>>>(s_zn, P(q_w), P(q_b), nullptr, s_a,
                                        128, 128, 0, 0, OUT_BF16_TM);
    gemm_kernel<<<gG, blk, 0, stream>>>(s_zn, P(k_w), P(k_b), nullptr, s_b,
                                        128, 128, 0, 0, OUT_BF16_TM);
    gemm_kernel<<<gG, blk, 0, stream>>>(s_zn, P(v_w), P(v_b), nullptr, s_g,
                                        128, 128, 0, 0, OUT_BF16_CM);
    gemm_kernel<<<dim3(512, 1), blk, 0, stream>>>(s_zn, P(bias_w), nullptr, nullptr,
                                                  s_tb, 128, 4, 0, 0, OUT_F32);
    tri_attn_kernel<<<8192, blk, 0, stream>>>((const __bf16*)s_a, (const __bf16*)s_b,
                                              (const __bf16*)s_g, s_tb, s_x);
    gemm_kernel<<<gG, blk, 0, stream>>>(s_zn, P(gate_w), P(gate_b), nullptr, s_xn,
                                        128, 128, 2, 0, OUT_F32);   // sigmoid
    mul_kernel<<<gEW4, blk, 0, stream>>>(s_xn, s_x, s_xt);          // g * o
    gemm_kernel<<<gG, blk, 0, stream>>>(s_xt, P(out_w), P(out_b), z, z,
                                        128, 128, 0, trans, OUT_F32);
  };
  attn_stage(TAS_LNB, TAS_LNG, TAS_QB, TAS_QW, TAS_KB, TAS_KW, TAS_VB, TAS_VW,
             TAS_BIASW, TAS_GATEB, TAS_GATEW, TAS_OUTB, TAS_OUTW, 0);
  attn_stage(TAE_LNB, TAE_LNG, TAE_QB, TAE_QW, TAE_KB, TAE_KW, TAE_VB, TAE_VW,
             TAE_BIASW, TAE_GATEB, TAE_GATEW, TAE_OUTB, TAE_OUTW, 1);

  // ---- pair transition ----
  ln_kernel<<<gLN, blk, 0, stream>>>(z, P(PT_LNG), P(PT_LNB), s_zn);
  gemm_kernel<<<dim3(512, 32), blk, 0, stream>>>(s_zn, P(PT_L1W), P(PT_L1B), nullptr,
                                                 s_big, 128, 512, 1, 0, OUT_F32);
  gemm_kernel<<<gG, blk, 0, stream>>>(s_big, P(PT_L2W), P(PT_L2B), z, z,
                                      512, 128, 0, 0, OUT_F32);
}
